// MambaBlock_88467736363293
// MI455X (gfx1250) — matrix operand order repaired and checked by reading, not compiled, not run on hardware
//
#include <hip/hip_runtime.h>

// ---------------------------------------------------------------------------
// Types for CDNA5 WMMA
// ---------------------------------------------------------------------------
typedef __bf16 bf16_t;
typedef __attribute__((ext_vector_type(16))) __bf16 v16bf;
typedef __attribute__((ext_vector_type(8)))  float  v8f;

union AFrag { v16bf v; unsigned int u[8]; };
union CAcc  { v8f   v; float        f[8]; };

#define D_MODEL 1024
#define SEQL    4096
#define BATCH   4
#define NSTATE  16
#define NTOK    (BATCH * SEQL)   // 16384

// ---------------------------------------------------------------------------
// CDNA5 async global->LDS copy (ASYNCcnt-tracked) + wait
// VDST operand = LDS byte address (low 32 bits of generic pointer to shared)
// ---------------------------------------------------------------------------
__device__ __forceinline__ unsigned lds_addr32(const void* p) {
    return (unsigned)(unsigned long long)p;
}
__device__ __forceinline__ void async_load_b128(unsigned lds_off, const void* gptr) {
    asm volatile("global_load_async_to_lds_b128 %0, %1, off"
                 :: "v"(lds_off), "v"(gptr) : "memory");
}
__device__ __forceinline__ void wait_asynccnt0() {
    asm volatile("s_wait_asynccnt 0x0" ::: "memory");
}

// ---------------------------------------------------------------------------
// Block-wide sum reduction helper (256 threads)
// ---------------------------------------------------------------------------
__device__ __forceinline__ float block_sum256(float v, float* red, int tid) {
    red[tid] = v; __syncthreads();
    for (int off = 128; off > 0; off >>= 1) {
        if (tid < off) red[tid] += red[tid + off];
        __syncthreads();
    }
    float r = red[0];
    __syncthreads();
    return r;
}

// ---------------------------------------------------------------------------
// Kernel 1: fused LN1 + depthwise causal conv (K=4) + residual
// ---------------------------------------------------------------------------
__global__ __launch_bounds__(256)
void conv_res_kernel(const float* __restrict__ x,
                     const float* __restrict__ ln1w, const float* __restrict__ ln1b,
                     const float* __restrict__ convw, const float* __restrict__ convb,
                     float* __restrict__ X1) {
    __shared__ float xn[11 * D_MODEL];
    __shared__ float red[256];
    const int tid = threadIdx.x;
    const int b   = blockIdx.x >> 9;            // SEQL/8 = 512 chunks per batch
    const int s0  = (blockIdx.x & 511) * 8;

    for (int r = 0; r < 11; ++r) {
        const int s = s0 - 3 + r;
        if (s < 0) {
            for (int j = 0; j < 4; ++j) xn[r * D_MODEL + j * 256 + tid] = 0.f;
            __syncthreads();
            continue;
        }
        const float* row = x + ((size_t)b * SEQL + s) * D_MODEL;
        float v[4], lsum = 0.f, lsq = 0.f;
        for (int j = 0; j < 4; ++j) {
            v[j] = row[j * 256 + tid];
            lsum += v[j]; lsq += v[j] * v[j];
        }
        const float mean = block_sum256(lsum, red, tid) * (1.f / D_MODEL);
        const float var  = block_sum256(lsq,  red, tid) * (1.f / D_MODEL) - mean * mean;
        const float inv  = rsqrtf(var + 1e-5f);
        for (int j = 0; j < 4; ++j) {
            const int c = j * 256 + tid;
            xn[r * D_MODEL + c] = (v[j] - mean) * inv * ln1w[c] + ln1b[c];
        }
        __syncthreads();
    }

    for (int j = 0; j < 4; ++j) {
        const int c = j * 256 + tid;
        const float w0 = convw[c * 4 + 0], w1 = convw[c * 4 + 1];
        const float w2 = convw[c * 4 + 2], w3 = convw[c * 4 + 3];
        const float cb = convb[c];
        for (int i = 0; i < 8; ++i) {
            const float acc = w0 * xn[(i + 0) * D_MODEL + c] + w1 * xn[(i + 1) * D_MODEL + c]
                            + w2 * xn[(i + 2) * D_MODEL + c] + w3 * xn[(i + 3) * D_MODEL + c];
            const size_t idx = ((size_t)b * SEQL + (s0 + i)) * D_MODEL + c;
            X1[idx] = x[idx] + acc + cb;
        }
    }
}

// ---------------------------------------------------------------------------
// Kernel 2: LayerNorm -> bf16 (one token per block)
// ---------------------------------------------------------------------------
__global__ __launch_bounds__(256)
void ln_bf16_kernel(const float* __restrict__ X, const float* __restrict__ w,
                    const float* __restrict__ bp, bf16_t* __restrict__ out) {
    __shared__ float red[256];
    const int tid = threadIdx.x;
    const size_t row = blockIdx.x;
    const float* xr = X + row * D_MODEL;
    float v[4], lsum = 0.f, lsq = 0.f;
    for (int j = 0; j < 4; ++j) {
        v[j] = xr[j * 256 + tid];
        lsum += v[j]; lsq += v[j] * v[j];
    }
    const float mean = block_sum256(lsum, red, tid) * (1.f / D_MODEL);
    const float var  = block_sum256(lsq,  red, tid) * (1.f / D_MODEL) - mean * mean;
    const float inv  = rsqrtf(var + 1e-5f);
    for (int j = 0; j < 4; ++j) {
        const int c = j * 256 + tid;
        out[row * D_MODEL + c] = (bf16_t)((v[j] - mean) * inv * w[c] + bp[c]);
    }
}

// ---------------------------------------------------------------------------
// Kernel 3: f32 -> bf16 weight convert
// ---------------------------------------------------------------------------
__global__ __launch_bounds__(256)
void f2bf_kernel(const float* __restrict__ in, bf16_t* __restrict__ out, int n) {
    const int i = blockIdx.x * 256 + threadIdx.x;
    if (i < n) out[i] = (bf16_t)in[i];
}

// ---------------------------------------------------------------------------
// Kernel 4: tiled bf16 WMMA GEMM, BM=128 BN=64 BK=32, 8 waves (wave32)
// double-buffered LDS; A tile staged with async global->LDS b128 copies,
// B tile staged via regs (transposed to [n][k]); one barrier per K-step.
// epilogues: 0 = split silu/sigmoid, 1 = exact gelu, 2 = +residual
// ---------------------------------------------------------------------------
enum { EPI_SPLIT = 0, EPI_GELU = 1, EPI_ADDRES = 2 };

#define SA_ELEMS (128 * 40)       // [m][k] bf16, +8 pad (per buffer)
#define SB_ELEMS (64 * 40)        // [n][k] bf16, +8 pad (per buffer)
#define SA_BYTES (SA_ELEMS * 2)

template<int EPI>
__global__ __launch_bounds__(256)
void gemm_bf16_kernel(const bf16_t* __restrict__ A, const bf16_t* __restrict__ B,
                      const float* __restrict__ bias, const float* __restrict__ res,
                      float* __restrict__ outF, bf16_t* __restrict__ out0,
                      bf16_t* __restrict__ out1, int M, int N, int K) {
    __shared__ __align__(16) unsigned short lds[2 * SA_ELEMS + 2 * SB_ELEMS];

    const int tid  = threadIdx.x;
    const int lane = tid & 31, wave = tid >> 5;
    const int wm = wave >> 1, wn = wave & 1;
    const int blkM = blockIdx.y * 128, blkN = blockIdx.x * 64;

    CAcc acc[2][2];
    for (int im = 0; im < 2; ++im)
        for (int in = 0; in < 2; ++in)
            for (int r = 0; r < 8; ++r) acc[im][in].f[r] = 0.f;

    const int arow = tid >> 1, achk = tid & 1;
    const int bkr  = tid >> 3, bnc  = tid & 7;
    const bf16_t* aRowG = A + (size_t)(blkM + arow) * K + achk * 16;
    // runtime LDS base (addrspacecast as an instruction, not a static init)
    const unsigned ldsBase = lds_addr32(lds);
    const unsigned aSlot   = (unsigned)((arow * 40 + achk * 16) * 2);  // byte off in A buf

    // ---- prologue: stage tile 0 into buffer 0 ----
    async_load_b128(ldsBase + aSlot, aRowG);
    {
        uint4 b0 = *(const uint4*)(B + (size_t)bkr * N + blkN + bnc * 8);
        const unsigned short* h = (const unsigned short*)&b0;
        unsigned short* sB0 = lds + 2 * SA_ELEMS;
        for (int i = 0; i < 8; ++i) sB0[(bnc * 8 + i) * 40 + bkr] = h[i];
    }
    wait_asynccnt0();
    __syncthreads();

    const int nK = K >> 5;
    for (int it = 0; it < nK; ++it) {
        const int cur = it & 1, nxt = cur ^ 1;
        const bool more = (it + 1) < nK;
        uint4 bReg;
        if (more) {   // issue next-tile loads before computing (latency hiding)
            const int kt = (it + 1) << 5;
            async_load_b128(ldsBase + (unsigned)nxt * SA_BYTES + aSlot, aRowG + kt);
            bReg = *(const uint4*)(B + (size_t)(kt + bkr) * N + blkN + bnc * 8);
        }

        const unsigned short* sAc = lds + cur * SA_ELEMS;
        const unsigned short* sBc = lds + 2 * SA_ELEMS + cur * SB_ELEMS;

        // A fragment: lanes 0-15 hold K=0..15, lanes 16-31 K=16..31 (M = lane&15)
        AFrag af[2], bfg[2];
        const int mrow0 = wm * 32 + (lane & 15);
        const int kbA   = (lane >> 4) * 16;
        for (int im = 0; im < 2; ++im) {
            const unsigned short* p = &sAc[(mrow0 + im * 16) * 40];
            for (int j = 0; j < 4; ++j) {
                af[im].u[j]     = *(const unsigned int*)&p[kbA + 2 * j];
                af[im].u[4 + j] = *(const unsigned int*)&p[kbA + 8 + 2 * j];
            }
        }
        // B fragment: lanes 0-15 hold K=0..15, lanes 16-31 K=16..31 (N = lane&15)
        const int nrow0 = wn * 32 + (lane & 15);
        const int kbB   = (lane >> 4) * 16;
        for (int in = 0; in < 2; ++in) {
            const unsigned short* p = &sBc[(nrow0 + in * 16) * 40];
            for (int j = 0; j < 8; ++j)
                bfg[in].u[j] = *(const unsigned int*)&p[kbB + 2 * j];
        }
        for (int im = 0; im < 2; ++im)
            for (int in = 0; in < 2; ++in)
                acc[im][in].v = __builtin_amdgcn_wmma_f32_16x16x32_bf16(
                    false, af[im].v, false, bfg[in].v, (short)0, acc[im][in].v,
                    false, false);

        if (more) {   // commit next B tile; drain async A writes
            unsigned short* sBn = lds + 2 * SA_ELEMS + nxt * SB_ELEMS;
            const unsigned short* h = (const unsigned short*)&bReg;
            for (int i = 0; i < 8; ++i) sBn[(bnc * 8 + i) * 40 + bkr] = h[i];
            wait_asynccnt0();
        }
        __syncthreads();
    }

    // epilogue (C layout: VGPR r -> M = r + 8*(lane>=16), N = lane&15)
    const int mBase = blkM + wm * 32 + ((lane >> 4) << 3);
    const int nBase = blkN + wn * 32 + (lane & 15);
    for (int im = 0; im < 2; ++im)
        for (int in = 0; in < 2; ++in) {
            const int n = nBase + in * 16;
            const float bi = bias[n];
            for (int r = 0; r < 8; ++r) {
                const int m = mBase + im * 16 + r;
                const float v = acc[im][in].f[r] + bi;
                if (EPI == EPI_SPLIT) {
                    const int h = N >> 1;
                    if (n < h)
                        out0[(size_t)m * h + n] = (bf16_t)(v / (1.f + __expf(-v)));
                    else
                        out1[(size_t)m * h + (n - h)] = (bf16_t)(1.f / (1.f + __expf(-v)));
                } else if (EPI == EPI_GELU) {
                    out0[(size_t)m * N + n] =
                        (bf16_t)(0.5f * v * (1.f + erff(v * 0.70710678118f)));
                } else {
                    outF[(size_t)m * N + n] = v + res[(size_t)m * N + n];
                }
            }
        }
}

// ---------------------------------------------------------------------------
// Kernel 5: skinny GEMM u = xp @ xp_w + b (N=16), double-buffered, async A
// ---------------------------------------------------------------------------
#define SBU_ELEMS (16 * 40)

__global__ __launch_bounds__(256)
void gemm_u_kernel(const bf16_t* __restrict__ A, const bf16_t* __restrict__ B,
                   const float* __restrict__ bias, float* __restrict__ U, int K) {
    __shared__ __align__(16) unsigned short lds[2 * SA_ELEMS + 2 * SBU_ELEMS];

    const int tid = threadIdx.x, lane = tid & 31, wave = tid >> 5;
    const int blkM = blockIdx.x * 128;
    CAcc acc; for (int r = 0; r < 8; ++r) acc.f[r] = 0.f;

    const int arow = tid >> 1, achk = tid & 1;
    const int bkr  = tid >> 1, bnc  = tid & 1;   // used only when tid < 64
    const bf16_t* aRowG = A + (size_t)(blkM + arow) * K + achk * 16;
    const unsigned ldsBase = lds_addr32(lds);
    const unsigned aSlot   = (unsigned)((arow * 40 + achk * 16) * 2);

    async_load_b128(ldsBase + aSlot, aRowG);
    if (tid < 64) {
        const uint4 d = *(const uint4*)(B + (size_t)bkr * NSTATE + bnc * 8);
        const unsigned short* h = (const unsigned short*)&d;
        unsigned short* sB0 = lds + 2 * SA_ELEMS;
        for (int i = 0; i < 8; ++i) sB0[(bnc * 8 + i) * 40 + bkr] = h[i];
    }
    wait_asynccnt0();
    __syncthreads();

    const int nK = K >> 5;
    for (int it = 0; it < nK; ++it) {
        const int cur = it & 1, nxt = cur ^ 1;
        const bool more = (it + 1) < nK;
        uint4 bReg;
        if (more) {
            const int kt = (it + 1) << 5;
            async_load_b128(ldsBase + (unsigned)nxt * SA_BYTES + aSlot, aRowG + kt);
            if (tid < 64)
                bReg = *(const uint4*)(B + (size_t)(kt + bkr) * NSTATE + bnc * 8);
        }

        const unsigned short* sAc = lds + cur * SA_ELEMS;
        const unsigned short* sBc = lds + 2 * SA_ELEMS + cur * SBU_ELEMS;

        AFrag af, bfg;
        const unsigned short* pa = &sAc[(wave * 16 + (lane & 15)) * 40];
        const int kbA = (lane >> 4) * 16;
        for (int j = 0; j < 4; ++j) {
            af.u[j]     = *(const unsigned int*)&pa[kbA + 2 * j];
            af.u[4 + j] = *(const unsigned int*)&pa[kbA + 8 + 2 * j];
        }
        const unsigned short* pb = &sBc[(lane & 15) * 40];
        const int kbB = (lane >> 4) * 16;
        for (int j = 0; j < 8; ++j)
            bfg.u[j] = *(const unsigned int*)&pb[kbB + 2 * j];
        acc.v = __builtin_amdgcn_wmma_f32_16x16x32_bf16(
            false, af.v, false, bfg.v, (short)0, acc.v, false, false);

        if (more) {
            if (tid < 64) {
                unsigned short* sBn = lds + 2 * SA_ELEMS + nxt * SBU_ELEMS;
                const unsigned short* h = (const unsigned short*)&bReg;
                for (int i = 0; i < 8; ++i) sBn[(bnc * 8 + i) * 40 + bkr] = h[i];
            }
            wait_asynccnt0();
        }
        __syncthreads();
    }

    const int n = lane & 15;
    const int mBase = blkM + wave * 16 + ((lane >> 4) << 3);
    for (int r = 0; r < 8; ++r)
        U[(size_t)(mBase + r) * NSTATE + n] = acc.f[r] + bias[n];
}

// ---------------------------------------------------------------------------
// Kernel 6: selective scan  h[t] = decay*h[t-1] + u[t]; one block per (b,n)
// chunked: 16 steps/thread, decay^16 carry chain. HS stored (s,b,n).
// ---------------------------------------------------------------------------
__global__ __launch_bounds__(256)
void scan_kernel(const float* __restrict__ U, const float* __restrict__ A_log,
                 float* __restrict__ HS) {
    __shared__ float carry[256];
    const int b = blockIdx.x >> 4, n = blockIdx.x & 15;
    const int t = threadIdx.x;
    const float decay = __expf(-__expf(A_log[n]));
    float u[16];
    const int sBase = t * 16;
    for (int i = 0; i < 16; ++i)
        u[i] = U[((size_t)b * SEQL + sBase + i) * NSTATE + n];
    float h = 0.f;
    for (int i = 0; i < 16; ++i) h = decay * h + u[i];
    carry[t] = h;
    __syncthreads();
    if (t == 0) {
        float d16 = 1.f;
        for (int i = 0; i < 16; ++i) d16 *= decay;
        float run = 0.f;
        for (int j = 0; j < 256; ++j) {
            const float c = carry[j];
            carry[j] = run;
            run = run * d16 + c;
        }
    }
    __syncthreads();
    float hh = carry[t];
    for (int i = 0; i < 16; ++i) {
        hh = decay * hh + u[i];
        HS[((size_t)(sBase + i) * BATCH + b) * NSTATE + n] = hh;
    }
}

// ---------------------------------------------------------------------------
// Kernel 7: Y[(s*B+b), d] = (hs . D[:,d]) * z[b,s,d]  -> bf16 (s,b,d layout)
// ---------------------------------------------------------------------------
__global__ __launch_bounds__(256)
void ydz_kernel(const float* __restrict__ HS, const float* __restrict__ Dm,
                const bf16_t* __restrict__ Z, bf16_t* __restrict__ Y) {
    __shared__ float hs[NSTATE];
    const int tokSB = blockIdx.x;          // = s*BATCH + b
    const int s = tokSB >> 2, b = tokSB & 3;
    const int tid = threadIdx.x;
    if (tid < NSTATE) hs[tid] = HS[(size_t)tokSB * NSTATE + tid];
    __syncthreads();
    for (int j = 0; j < 4; ++j) {
        const int c = j * 256 + tid;
        float a = 0.f;
        for (int nn = 0; nn < NSTATE; ++nn) a += hs[nn] * Dm[nn * D_MODEL + c];
        const float z = (float)Z[((size_t)b * SEQL + s) * D_MODEL + c];
        Y[(size_t)tokSB * D_MODEL + c] = (bf16_t)(a * z);
    }
}

// ---------------------------------------------------------------------------
// Host orchestration
// ---------------------------------------------------------------------------
extern "C" void kernel_launch(void* const* d_in, const int* in_sizes, int n_in,
                              void* d_out, int out_size, void* d_ws, size_t ws_size,
                              hipStream_t stream) {
    (void)in_sizes; (void)n_in; (void)out_size; (void)ws_size;
    const float* x      = (const float*)d_in[0];
    const float* ln1w   = (const float*)d_in[1];
    const float* ln1b   = (const float*)d_in[2];
    const float* convw  = (const float*)d_in[3];
    const float* convb  = (const float*)d_in[4];
    const float* lnsw   = (const float*)d_in[5];
    const float* lnsb   = (const float*)d_in[6];
    const float* in_w   = (const float*)d_in[7];
    const float* in_b   = (const float*)d_in[8];
    const float* xp_w   = (const float*)d_in[9];
    const float* xp_b   = (const float*)d_in[10];
    const float* A_log  = (const float*)d_in[11];
    const float* Dmat   = (const float*)d_in[12];
    const float* out_w  = (const float*)d_in[13];
    const float* out_b  = (const float*)d_in[14];
    const float* ln2w   = (const float*)d_in[15];
    const float* ln2b   = (const float*)d_in[16];
    const float* w1     = (const float*)d_in[17];
    const float* b1     = (const float*)d_in[18];
    const float* w2     = (const float*)d_in[19];
    const float* b2     = (const float*)d_in[20];
    float* outp = (float*)d_out;

    // ---- workspace carve-up (256B aligned) ----
    char* ws = (char*)d_ws;
    size_t off = 0;
    auto alloc = [&](size_t bytes) {
        char* p = ws + off;
        off += (bytes + 255) & ~(size_t)255;
        return p;
    };
    float*  X1   = (float*) alloc((size_t)NTOK * D_MODEL * 4);   // conv+res out (reused as H1 bf16)
    bf16_t* BUFB = (bf16_t*)alloc((size_t)NTOK * D_MODEL * 2);   // XN -> Y -> H
    bf16_t* XP   = (bf16_t*)alloc((size_t)NTOK * D_MODEL * 2);
    bf16_t* Z    = (bf16_t*)alloc((size_t)NTOK * D_MODEL * 2);
    float*  U    = (float*) alloc((size_t)NTOK * NSTATE * 4);
    float*  HS   = (float*) alloc((size_t)NTOK * NSTATE * 4);
    float*  X2   = (float*) alloc((size_t)NTOK * D_MODEL * 4);
    bf16_t* inw_bf = (bf16_t*)alloc((size_t)D_MODEL * 2 * D_MODEL * 2);
    bf16_t* xpw_bf = (bf16_t*)alloc((size_t)D_MODEL * NSTATE * 2);
    bf16_t* outw_bf= (bf16_t*)alloc((size_t)D_MODEL * D_MODEL * 2);
    bf16_t* w1_bf  = (bf16_t*)alloc((size_t)D_MODEL * 2 * D_MODEL * 2);
    bf16_t* w2_bf  = (bf16_t*)alloc((size_t)2 * D_MODEL * D_MODEL * 2);
    bf16_t* H1 = (bf16_t*)X1;   // reuse X1 slab after its residual is consumed

    // ---- weight converts ----
    f2bf_kernel<<<(D_MODEL * 2 * D_MODEL + 255) / 256, 256, 0, stream>>>(in_w, inw_bf, D_MODEL * 2 * D_MODEL);
    f2bf_kernel<<<(D_MODEL * NSTATE + 255) / 256, 256, 0, stream>>>(xp_w, xpw_bf, D_MODEL * NSTATE);
    f2bf_kernel<<<(D_MODEL * D_MODEL + 255) / 256, 256, 0, stream>>>(out_w, outw_bf, D_MODEL * D_MODEL);
    f2bf_kernel<<<(D_MODEL * 2 * D_MODEL + 255) / 256, 256, 0, stream>>>(w1, w1_bf, D_MODEL * 2 * D_MODEL);
    f2bf_kernel<<<(2 * D_MODEL * D_MODEL + 255) / 256, 256, 0, stream>>>(w2, w2_bf, 2 * D_MODEL * D_MODEL);

    // ---- 1. LN1 + depthwise causal conv + residual ----
    conv_res_kernel<<<NTOK / 8, 256, 0, stream>>>(x, ln1w, ln1b, convw, convb, X1);

    // ---- 2. LN_s -> bf16 ----
    ln_bf16_kernel<<<NTOK, 256, 0, stream>>>(X1, lnsw, lnsb, BUFB);

    // ---- 3. in-proj GEMM (16384x1024 @ 1024x2048) + silu/sigmoid split ----
    gemm_bf16_kernel<EPI_SPLIT><<<dim3(2 * D_MODEL / 64, NTOK / 128), 256, 0, stream>>>(
        BUFB, inw_bf, in_b, nullptr, nullptr, XP, Z, NTOK, 2 * D_MODEL, D_MODEL);

    // ---- 4. u = xp @ xp_w + b (N=16 skinny) ----
    gemm_u_kernel<<<NTOK / 128, 256, 0, stream>>>(XP, xpw_bf, xp_b, U, D_MODEL);

    // ---- 5. selective scan ----
    scan_kernel<<<BATCH * NSTATE, 256, 0, stream>>>(U, A_log, HS);

    // ---- 6. ys = (hs @ D) * z, stored (s,b,d) -> reuse BUFB ----
    ydz_kernel<<<NTOK, 256, 0, stream>>>(HS, Dmat, Z, BUFB);

    // ---- 7. out GEMM + bias + residual (faithful flat reshape) ----
    gemm_bf16_kernel<EPI_ADDRES><<<dim3(D_MODEL / 64, NTOK / 128), 256, 0, stream>>>(
        BUFB, outw_bf, out_b, X1, X2, nullptr, nullptr, NTOK, D_MODEL, D_MODEL);

    // ---- 8. LN2 -> bf16 (reuse BUFB) ----
    ln_bf16_kernel<<<NTOK, 256, 0, stream>>>(X2, ln2w, ln2b, BUFB);

    // ---- 9. MLP up GEMM + exact GELU (H1 reuses X1 slab) ----
    gemm_bf16_kernel<EPI_GELU><<<dim3(2 * D_MODEL / 64, NTOK / 128), 256, 0, stream>>>(
        BUFB, w1_bf, b1, nullptr, nullptr, H1, nullptr, NTOK, 2 * D_MODEL, D_MODEL);

    // ---- 10. MLP down GEMM + bias + residual -> output ----
    gemm_bf16_kernel<EPI_ADDRES><<<dim3(D_MODEL / 64, NTOK / 128), 256, 0, stream>>>(
        H1, w2_bf, b2, X2, outp, nullptr, nullptr, NTOK, D_MODEL, 2 * D_MODEL);
}
